// GroupedQueryAttention_37933151158284
// MI455X (gfx1250) — matrix operand order repaired and checked by reading, not compiled, not run on hardware
//
#include <hip/hip_runtime.h>
#include <hip/hip_bf16.h>

// ---------------------------------------------------------------------------
// GQA for MI455X (gfx1250): all matmuls via v_wmma_f32_16x16x32_bf16 (wave32).
// Global->LDS staging: per-lane GLOBAL_LOAD_ASYNC_TO_LDS_B128 (ASYNCcnt) for
// GEMM A tiles; TENSOR_LOAD_TO_LDS (TDM, TENSORcnt) for GEMM weight tiles and
// attention K tiles. Double-buffered everywhere DMA latency matters.
// Softmax row-sums computed by WMMA against a ones matrix.
// ---------------------------------------------------------------------------

typedef __attribute__((ext_vector_type(16))) __bf16 v16bf;
typedef __attribute__((ext_vector_type(8)))  __bf16 bf16x8;
typedef __attribute__((ext_vector_type(4)))  __bf16 bf16x4;
typedef __attribute__((ext_vector_type(2)))  __bf16 bf16x2;
typedef __attribute__((ext_vector_type(8)))  float  v8f;
typedef __attribute__((ext_vector_type(4)))  unsigned u32x4;
typedef __attribute__((ext_vector_type(8)))  unsigned u32x8;

#define WMMA_BF16(a, b, c) \
    __builtin_amdgcn_wmma_f32_16x16x32_bf16(false, (a), false, (b), (short)0, (c), false, false)

union Frag16 { v16bf v; bf16x8 h[2]; };

static __device__ __forceinline__ v16bf ld_frag(const __bf16* lo, const __bf16* hi) {
    Frag16 f;
    f.h[0] = *(const bf16x8*)lo;
    f.h[1] = *(const bf16x8*)hi;
    return f.v;
}

// 32-bit LDS address of a shared-memory pointer (flat aperture: addr[31:0])
static __device__ __forceinline__ unsigned lds_off(const void* p) {
    return (unsigned)(unsigned long long)p;
}

// Async DMA: 16B from (base + voff_bytes) -> LDS[dst]; tracked by ASYNCcnt.
static __device__ __forceinline__ void async_b128(unsigned dst, const void* base, unsigned voff) {
    asm volatile("global_load_async_to_lds_b128 %0, %1, %2"
                 :: "v"(dst), "v"(voff), "s"((unsigned long long)base)
                 : "memory");
}
static __device__ __forceinline__ void wait_async0() {
    asm volatile("s_wait_asynccnt 0" ::: "memory");
}

// TDM: DMA a 2D tile (tile0 x tile1 elements of 2B, row stride strideElems)
// from global memory into LDS, padding padAmt(+1) DWORDs after every
// 2^(padIv+1) DWORDs. One issue per wave; tracked by TENSORcnt.
static __device__ __forceinline__ void tdm_load_tile_2d(unsigned ldsAddr,
                                                        unsigned long long gaddr,
                                                        unsigned tile0, unsigned tile1,
                                                        unsigned long long strideElems,
                                                        unsigned padIv, unsigned padAmt) {
    u32x4 g0;
    g0[0] = 1u;                                      // count=1, user mode
    g0[1] = ldsAddr;                                 // lds_addr (bytes)
    g0[2] = (unsigned)gaddr;                         // global_addr[31:0]
    g0[3] = (unsigned)(gaddr >> 32) | (2u << 30);    // global_addr[56:32] | type=2
    const unsigned td0 = 1u << 20, td1 = 1u << 20;   // tensor dims (no OOB clipping)
    u32x8 g1;
    g1[0] = (1u << 16)        // data_size = 2 bytes
          | (1u << 20)        // pad_enable
          | (padIv << 22)     // pad_interval
          | (padAmt << 25);   // pad_amount
    g1[1] = (td0 & 0xFFFFu) << 16;                               // tensor_dim0 lo
    g1[2] = ((td0 >> 16) & 0xFFFFu) | ((td1 & 0xFFFFu) << 16);   // dim0 hi | dim1 lo
    g1[3] = ((td1 >> 16) & 0xFFFFu) | (tile0 << 16);             // dim1 hi | tile_dim0
    g1[4] = tile1;                                               // tile_dim1 | tile_dim2=0
    g1[5] = (unsigned)(strideElems & 0xFFFFFFFFu);               // dim0_stride lo
    g1[6] = (unsigned)((strideElems >> 32) & 0xFFFFu);           // dim0_stride hi
    g1[7] = 0u;
    asm volatile("tensor_load_to_lds %0, %1" :: "s"(g0), "s"(g1) : "memory");
}

// ---------------------------------------------------------------------------
// f32 -> bf16 elementwise convert (n multiple of 4)
// ---------------------------------------------------------------------------
__global__ __launch_bounds__(256) void cvt_bf16_kernel(const float* __restrict__ in,
                                                       __bf16* __restrict__ out, int n) {
    int i = (blockIdx.x * 256 + threadIdx.x) * 4;
    if (i < n) {
        float4 v = *(const float4*)(in + i);
        bf16x4 o;
        o[0] = (__bf16)v.x; o[1] = (__bf16)v.y; o[2] = (__bf16)v.z; o[3] = (__bf16)v.w;
        *(bf16x4*)(out + i) = o;
    }
}

// ---------------------------------------------------------------------------
// Weight transpose + convert: Wt[n*K + k] = (bf16) W[k*N + n]
// grid: (K/32, N/8), block 256 (= 32 x 8)
// ---------------------------------------------------------------------------
__global__ __launch_bounds__(256) void transpose_bf16_kernel(const float* __restrict__ W,
                                                             __bf16* __restrict__ Wt,
                                                             int K, int N) {
    int k = blockIdx.x * 32 + (threadIdx.x & 31);
    int n = blockIdx.y * 8 + (threadIdx.x >> 5);
    if (k < K && n < N)
        Wt[(long)n * K + k] = (__bf16)W[(long)k * N + n];
}

// ---------------------------------------------------------------------------
// Generic GEMM: C[M,N] = A[M,K](bf16) * Wt[N,K](bf16)^T + bias[N]
// Block tile 128x128, 256 threads = 8 waves (4 along M x 2 along N),
// each wave 32x64 = 2x4 WMMA 16x16 tiles. K-step 32. Double-buffered LDS:
// A tile by per-lane async DMA, B tile by one TDM descriptor (wave 0),
// both overlapped with WMMA compute.
// LDS row stride 40 bf16 (80B = 5x16B) -> conflict-free b128 fragment loads.
// ---------------------------------------------------------------------------
template <typename OutT>
__global__ __launch_bounds__(256) void gemm_wmma_kernel(const __bf16* __restrict__ A,
                                                        const __bf16* __restrict__ Wt,
                                                        const float* __restrict__ bias,
                                                        OutT* __restrict__ C,
                                                        int M, int N, int K) {
    __shared__ __bf16 sA[2][128 * 40];
    __shared__ __bf16 sB[2][128 * 40];

    const int t = threadIdx.x;
    const int lane = t & 31, w = t >> 5;
    const int wm = w & 3, wn = w >> 2;            // 4 M-waves x 2 N-waves
    const int m0 = blockIdx.y * 128, n0 = blockIdx.x * 128;
    const int laneRow = lane & 15;
    const int kA = (lane >> 4) * 8;               // unused: A fragments also use kB
    const int kB = (lane >> 4) * 16;              // B-fragment K base

    // this thread's two 16B A-staging chunks: rows r, r+64, same 8-col group
    const int r = t >> 2, c8 = (t & 3) << 3;
    const int nk = K >> 5;

    const v8f z = {0.f, 0.f, 0.f, 0.f, 0.f, 0.f, 0.f, 0.f};
    v8f acc[2][4];
#pragma unroll
    for (int mt = 0; mt < 2; ++mt)
#pragma unroll
        for (int nt = 0; nt < 4; ++nt) acc[mt][nt] = z;

    // stage one 128x32 A tile (async) + 128x32 B tile (TDM) into buffer `buf`
    auto stage = [&](int buf, int k0) {
        unsigned gA = (unsigned)(((long)(m0 + r) * K + k0 + c8) * 2);
        unsigned stride = (unsigned)(64L * K * 2);
        unsigned lA = lds_off(&sA[buf][r * 40 + c8]);
        async_b128(lA, A, gA);
        async_b128(lA + 64u * 40u * 2u, A, gA + stride);
        if (w == 0) {
            unsigned long long gB = (unsigned long long)Wt + ((long)n0 * K + k0) * 2;
            // 64B rows (16 DW -> code 3), 16B pad (4 DW -> code 3) => 80B pitch
            tdm_load_tile_2d(lds_off(&sB[buf][0]), gB, 32u, 128u,
                             (unsigned long long)K, 3u, 3u);
        }
    };
    auto drain = [&]() {
        wait_async0();
        if (w == 0) __builtin_amdgcn_s_wait_tensorcnt(0);
    };

    stage(0, 0);
    drain();
    __syncthreads();

    for (int i = 0; i < nk; ++i) {
        const int cur = i & 1;
        if (i + 1 < nk) stage(cur ^ 1, (i + 1) << 5);   // overlap DMA with WMMA

        v16bf af[2], bfrag[4];
#pragma unroll
        for (int mt = 0; mt < 2; ++mt) {
            const __bf16* p = &sA[cur][(wm * 32 + mt * 16 + laneRow) * 40];
            af[mt] = ld_frag(p + kB, p + 8 + kB);
        }
#pragma unroll
        for (int nt = 0; nt < 4; ++nt) {
            const __bf16* p = &sB[cur][(wn * 64 + nt * 16 + laneRow) * 40 + kB];
            bfrag[nt] = ld_frag(p, p + 8);
        }
#pragma unroll
        for (int mt = 0; mt < 2; ++mt)
#pragma unroll
            for (int nt = 0; nt < 4; ++nt)
                acc[mt][nt] = WMMA_BF16(af[mt], bfrag[nt], acc[mt][nt]);

        if (i + 1 < nk) drain();
        __syncthreads();
    }

    // ---- epilogue: C layout row = i + 8*(lane>=16), col = lane&15
    const int rbase = m0 + wm * 32 + ((lane >> 4) << 3);
#pragma unroll
    for (int nt = 0; nt < 4; ++nt) {
        int col = n0 + wn * 64 + nt * 16 + laneRow;
        float bv = bias[col];
#pragma unroll
        for (int mt = 0; mt < 2; ++mt)
#pragma unroll
            for (int i = 0; i < 8; ++i)
                C[(long)(rbase + mt * 16 + i) * N + col] = (OutT)(acc[mt][nt][i] + bv);
    }
}

// ---------------------------------------------------------------------------
// Flash attention, one (b,g,q) head per blockIdx.y, 128 query rows per block.
// 256 threads = 8 waves, each wave owns 16 query rows. Key blocks of 64.
// Double-buffered staging: K via TDM descriptor (wave 0), V loaded to regs
// at loop top and transposed into LDS after compute -> DMA latency hidden
// behind WMMA + softmax. Row-sums via WMMA against a ones matrix.
// ---------------------------------------------------------------------------
__global__ __launch_bounds__(256) void gqa_attn_kernel(const __bf16* __restrict__ QH,
                                                       const __bf16* __restrict__ KH,
                                                       const __bf16* __restrict__ VH,
                                                       __bf16* __restrict__ AO) {
    const int S = 2048, Gc = 4, Qc = 4, D = 64;
    __shared__ __bf16 sK[2][64 * 72];   // [key][d]
    __shared__ __bf16 sVt[2][64 * 72];  // [d][key]  (transposed)
    __shared__ __bf16 sP[8][16 * 72];   // per-wave P staging

    const int t = threadIdx.x, lane = t & 31, w = t >> 5;
    const int laneRow = lane & 15;
    const int kA = (lane >> 4) * 8;     // unused: A fragments also use kB
    const int kB = (lane >> 4) * 16;    // B-fragment K base
    const int rowOff = (lane >> 4) << 3;

    const int head = blockIdx.y;
    const int b = head >> 4, hr = head & 15, g = hr >> 2, q = hr & 3;
    const int qbase = blockIdx.x * 128 + w * 16;

    // V staging coords: 256 key-pair chunks (2 keys x 8 d), one per thread.
    const int kp = t >> 3, ck8 = (t & 7) << 3;

    // Q fragments for this wave's 16 rows (loaded once, straight from global)
    const __bf16* qrow = QH + ((long)(b * S + qbase + laneRow) * (Gc * Qc) + g * Qc + q) * D;
    v16bf qf[2];
#pragma unroll
    for (int kc = 0; kc < 2; ++kc)
        qf[kc] = ld_frag(qrow + kc * 32 + kB, qrow + kc * 32 + 8 + kB);

    // all-ones B matrix (32x16) for WMMA row-sum
    v16bf onesb;
#pragma unroll
    for (int j = 0; j < 16; ++j) onesb[j] = (__bf16)1.0f;

    const v8f z = {0.f, 0.f, 0.f, 0.f, 0.f, 0.f, 0.f, 0.f};
    v8f o[4];
    float mrun[8], lrun[8];
#pragma unroll
    for (int dt = 0; dt < 4; ++dt) o[dt] = z;
#pragma unroll
    for (int i = 0; i < 8; ++i) { mrun[i] = -1e30f; lrun[i] = 0.f; }

    const float scale = 1.0f / 32.0f;   // 1/sqrt(E), E=1024

    // ---- staging helpers
    auto stageK = [&](int buf, int t0) {
        if (w == 0) {
            unsigned long long gK = (unsigned long long)KH +
                (((long)(b * S + t0) * Gc + g) * D) * 2;
            // 128B rows (32 DW -> code 4), 16B pad (4 DW -> code 3) => 144B pitch
            tdm_load_tile_2d(lds_off(&sK[buf][0]), gK, 64u, 64u,
                             (unsigned long long)(Gc * D), 4u, 3u);
        }
    };
    auto loadV = [&](int t0, bf16x8& va, bf16x8& vb) {
        long src = ((long)(b * S + t0 + 2 * kp) * Gc + g) * D + ck8;
        va = *(const bf16x8*)&VH[src];
        vb = *(const bf16x8*)&VH[src + Gc * D];
    };
    auto storeV = [&](int buf, bf16x8 va, bf16x8 vb) {
#pragma unroll
        for (int j = 0; j < 8; ++j) {
            bf16x2 p; p[0] = va[j]; p[1] = vb[j];
            *(bf16x2*)&sVt[buf][(ck8 + j) * 72 + 2 * kp] = p;
        }
    };

    // ---- prologue: stage block 0
    bf16x8 va, vb;
    stageK(0, 0);
    loadV(0, va, vb);
    storeV(0, va, vb);
    if (w == 0) __builtin_amdgcn_s_wait_tensorcnt(0);
    __syncthreads();

    for (int it = 0; it < S / 64; ++it) {
        const int cur = it & 1;
        const int t0n = (it + 1) * 64;
        const bool more = t0n < S;
        if (more) {                     // kick off next block's DMA + V loads
            stageK(cur ^ 1, t0n);
            loadV(t0n, va, vb);
        }

        // ---- scores S16x64 = Q(16x64) . K^T
        v8f sc[4];
#pragma unroll
        for (int nt = 0; nt < 4; ++nt) {
            v8f s = z;
            const __bf16* kpnt = &sK[cur][(nt * 16 + laneRow) * 72 + kB];
            s = WMMA_BF16(qf[0], ld_frag(kpnt, kpnt + 8), s);
            s = WMMA_BF16(qf[1], ld_frag(kpnt + 32, kpnt + 40), s);
            sc[nt] = s;
        }

        // ---- online softmax: row max via shuffles, P = exp(s*scale - m)
        float corr[8];
#pragma unroll
        for (int i = 0; i < 8; ++i) {
            float rm = -1e30f;
#pragma unroll
            for (int nt = 0; nt < 4; ++nt) rm = fmaxf(rm, sc[nt][i]);
            rm = fmaxf(rm, __shfl_xor(rm, 1));
            rm = fmaxf(rm, __shfl_xor(rm, 2));
            rm = fmaxf(rm, __shfl_xor(rm, 4));
            rm = fmaxf(rm, __shfl_xor(rm, 8));
            rm *= scale;
            float mnew = fmaxf(mrun[i], rm);
            corr[i] = __expf(mrun[i] - mnew);
            mrun[i] = mnew;
#pragma unroll
            for (int nt = 0; nt < 4; ++nt)
                sc[nt][i] = __expf(sc[nt][i] * scale - mnew);
#pragma unroll
            for (int dt = 0; dt < 4; ++dt) o[dt][i] *= corr[i];
        }

        // ---- P: C-layout -> A-layout via per-wave LDS (same-wave DS in-order)
#pragma unroll
        for (int i = 0; i < 8; ++i)
#pragma unroll
            for (int nt = 0; nt < 4; ++nt)
                sP[w][(rowOff + i) * 72 + nt * 16 + laneRow] = (__bf16)sc[nt][i];

        v16bf pf[2];
        const __bf16* pp = &sP[w][laneRow * 72];
        pf[0] = ld_frag(pp + kB, pp + 8 + kB);
        pf[1] = ld_frag(pp + 32 + kB, pp + 40 + kB);

        // ---- row sums of P via WMMA with ones matrix (every col = rowsum)
        v8f rs = z;
        rs = WMMA_BF16(pf[0], onesb, rs);
        rs = WMMA_BF16(pf[1], onesb, rs);
#pragma unroll
        for (int i = 0; i < 8; ++i) lrun[i] = lrun[i] * corr[i] + rs[i];

        // ---- O += P(16x64) . V(64x64)
#pragma unroll
        for (int dt = 0; dt < 4; ++dt) {
            const __bf16* vp = &sVt[cur][(dt * 16 + laneRow) * 72 + kB];
            o[dt] = WMMA_BF16(pf[0], ld_frag(vp, vp + 8), o[dt]);
            o[dt] = WMMA_BF16(pf[1], ld_frag(vp + 32, vp + 40), o[dt]);
        }

        if (more) {                     // commit next block's V, drain K DMA
            storeV(cur ^ 1, va, vb);
            if (w == 0) __builtin_amdgcn_s_wait_tensorcnt(0);
        }
        __syncthreads();
    }

    // ---- normalize and store (bf16, layout [B,S,(g*Q+q)*D + d])
#pragma unroll
    for (int dt = 0; dt < 4; ++dt)
#pragma unroll
        for (int i = 0; i < 8; ++i) {
            int row = qbase + rowOff + i;
            int d = dt * 16 + laneRow;
            float val = o[dt][i] / lrun[i];
            AO[((long)(b * S + row) * (Gc * Qc) + g * Qc + q) * D + d] = (__bf16)val;
        }
}

// ---------------------------------------------------------------------------
// Host-side orchestration
// ---------------------------------------------------------------------------
extern "C" void kernel_launch(void* const* d_in, const int* in_sizes, int n_in,
                              void* d_out, int out_size, void* d_ws, size_t ws_size,
                              hipStream_t stream) {
    (void)in_sizes; (void)n_in; (void)out_size; (void)ws_size;
    const int B = 2, S = 2048, E = 1024, G = 4, Q = 4, D = 64;
    const int M = B * S;          // 4096 tokens
    const int GQD = G * Q * D;    // 1024
    const int GD = G * D;         // 256

    const float* q    = (const float*)d_in[0];
    const float* k    = (const float*)d_in[1];
    const float* v    = (const float*)d_in[2];
    const float* Wq_w = (const float*)d_in[3];
    const float* Wq_b = (const float*)d_in[4];
    const float* Wk_w = (const float*)d_in[5];
    const float* Wk_b = (const float*)d_in[6];
    const float* Wv_w = (const float*)d_in[7];
    const float* Wv_b = (const float*)d_in[8];
    const float* fc_w = (const float*)d_in[9];
    const float* fc_b = (const float*)d_in[10];
    float* out = (float*)d_out;

    // workspace carve-out (bf16 buffers), 256B aligned
    char* ws = (char*)d_ws;
    size_t off = 0;
    auto carve = [&](size_t elems) {
        void* p = ws + off;
        off = (off + elems * sizeof(__bf16) + 255) & ~(size_t)255;
        return (__bf16*)p;
    };
    __bf16* qbf = carve((size_t)M * E);
    __bf16* kbf = carve((size_t)M * E);
    __bf16* vbf = carve((size_t)M * E);
    __bf16* WqT = carve((size_t)E * GQD);
    __bf16* WkT = carve((size_t)E * GD);
    __bf16* WvT = carve((size_t)E * GD);
    __bf16* fcT = carve((size_t)GQD * GQD);
    __bf16* QHb = carve((size_t)M * GQD);
    __bf16* KHb = carve((size_t)M * GD);
    __bf16* VHb = carve((size_t)M * GD);
    __bf16* AOb = carve((size_t)M * GQD);

    // 1) activations -> bf16
    {
        int n = M * E;
        dim3 grid(n / (4 * 256));
        cvt_bf16_kernel<<<grid, 256, 0, stream>>>(q, qbf, n);
        cvt_bf16_kernel<<<grid, 256, 0, stream>>>(k, kbf, n);
        cvt_bf16_kernel<<<grid, 256, 0, stream>>>(v, vbf, n);
    }
    // 2) weights -> transposed bf16 (N-major)
    transpose_bf16_kernel<<<dim3(E / 32, GQD / 8), 256, 0, stream>>>(Wq_w, WqT, E, GQD);
    transpose_bf16_kernel<<<dim3(E / 32, GD / 8), 256, 0, stream>>>(Wk_w, WkT, E, GD);
    transpose_bf16_kernel<<<dim3(E / 32, GD / 8), 256, 0, stream>>>(Wv_w, WvT, E, GD);
    transpose_bf16_kernel<<<dim3(GQD / 32, GQD / 8), 256, 0, stream>>>(fc_w, fcT, GQD, GQD);

    // 3) projections (WMMA GEMM, bf16 out)
    gemm_wmma_kernel<__bf16><<<dim3(GQD / 128, M / 128), 256, 0, stream>>>(qbf, WqT, Wq_b, QHb, M, GQD, E);
    gemm_wmma_kernel<__bf16><<<dim3(GD / 128, M / 128), 256, 0, stream>>>(kbf, WkT, Wk_b, KHb, M, GD, E);
    gemm_wmma_kernel<__bf16><<<dim3(GD / 128, M / 128), 256, 0, stream>>>(vbf, WvT, Wv_b, VHb, M, GD, E);

    // 4) attention (flash, WMMA)
    gqa_attn_kernel<<<dim3(S / 128, B * G * Q), 256, 0, stream>>>(QHb, KHb, VHb, AOb);

    // 5) output projection (WMMA GEMM, f32 out)
    gemm_wmma_kernel<float><<<dim3(GQD / 128, M / 128), 256, 0, stream>>>(AOb, fcT, fc_b, out, M, GQD, GQD);
}
